// CausalSelfAttention_16793322127460
// MI455X (gfx1250) — matrix operand order repaired and checked by reading, not compiled, not run on hardware
//
#include <hip/hip_runtime.h>
#include <stdint.h>

// ---------------------------------------------------------------------------
// Types
// ---------------------------------------------------------------------------
typedef __attribute__((ext_vector_type(16))) __bf16        v16bf;
typedef __attribute__((ext_vector_type(8)))  float         v8f;
typedef __attribute__((ext_vector_type(4)))  unsigned int  u32x4;

// async-to-LDS builtin operand types: vector_size(16) int, AS1 src / AS3 dst
typedef int v4i_t __attribute__((vector_size(16)));
typedef __attribute__((address_space(1))) v4i_t* gv4i_p;   // global int4*
typedef __attribute__((address_space(3))) v4i_t* lv4i_p;   // LDS int4*

#if defined(__has_builtin)
#if __has_builtin(__builtin_amdgcn_global_load_async_to_lds_b128)
#define ASYNC_LDS 1
#endif
#endif
#ifndef ASYNC_LDS
#define ASYNC_LDS 0
#endif

#define B_   4
#define T_   2048
#define C_   1024
#define H_   16
#define D_   64
#define NTOK (B_ * T_)   // 8192 tokens
#define N3C  (3 * C_)    // 3072

__device__ __forceinline__ unsigned short f2bf(float x) {
  unsigned u = __builtin_bit_cast(unsigned, x);
  u += 0x7FFFu + ((u >> 16) & 1u);          // round-to-nearest-even
  return (unsigned short)(u >> 16);
}

union Frag {
  v16bf v;
  u32x4 q[2];
};

// A-matrix fragment (16x32, 16-bit), source row-major [M][K], K-chunk = 32.
// lanes 0-15: M=lane, K = 0..7 (V0-3) and 16..23 (V4-7)
// lanes 16-31: M=lane-16, K = 8..15 and 24..31
__device__ __forceinline__ void load_a_frag(Frag& f, const unsigned short* base, int ld) {
  const int lane = threadIdx.x & 31;
  const int m  = lane & 15;
  const int hi = lane >> 4;
  const unsigned short* p = base + (size_t)m * ld + hi * 8;
  f.q[0] = *reinterpret_cast<const u32x4*>(p);
  f.q[1] = *reinterpret_cast<const u32x4*>(p + 16);
}

// B-matrix fragment (32x16, 16-bit), source stored N-major: [N][K], K-chunk = 32.
// lanes 0-15: N=lane, K = 0..7 and 16..23 ; lanes 16-31: N=lane-16, K = 8..15 and 24..31
__device__ __forceinline__ void load_b_frag(Frag& f, const unsigned short* base, int ld) {
  const int lane = threadIdx.x & 31;
  const int n  = lane & 15;
  const int hi = lane >> 4;
  const unsigned short* p = base + (size_t)n * ld + hi * 8;
  f.q[0] = *reinterpret_cast<const u32x4*>(p);
  f.q[1] = *reinterpret_cast<const u32x4*>(p + 16);
}

__device__ __forceinline__ v8f wmma_bf16(const Frag& a, const Frag& b, v8f c) {
  return __builtin_amdgcn_wmma_f32_16x16x32_bf16(
      /*neg_a=*/false, a.v, /*neg_b=*/false, b.v,
      /*c_mod=*/(short)0, c, /*reuse_a=*/false, /*reuse_b=*/false);
}

// ---------------------------------------------------------------------------
// Cast / transpose-cast kernels (fp32 -> bf16-as-u16)
// ---------------------------------------------------------------------------
__global__ void k_cast(const float* __restrict__ src, unsigned short* __restrict__ dst, int n) {
  int i = blockIdx.x * blockDim.x + threadIdx.x;
  if (i < n) dst[i] = f2bf(src[i]);
}

// src[rows][cols] (fp32) -> dst[cols][rows] (bf16)
__global__ void k_transpose_cast(const float* __restrict__ src, unsigned short* __restrict__ dst,
                                 int rows, int cols) {
  int c = blockIdx.x * 16 + threadIdx.x;
  int r = blockIdx.y * 16 + threadIdx.y;
  if (r < rows && c < cols)
    dst[(size_t)c * rows + r] = f2bf(src[(size_t)r * cols + c]);
}

// ---------------------------------------------------------------------------
// 128x128 workgroup GEMM core: C(128x128) += A(128xK) * Bt(128xK)^T
// 4 waves in 2x2, each wave owns 64x64 (4x4 WMMA accumulators).
// A, Bt row strides == Kdim.  Staging uses the CDNA5 async direct-to-LDS
// path (GLOBAL_LOAD_ASYNC_TO_LDS_B128 / ASYNCcnt) when available: the same
// instruction offset is applied to both the global and LDS address, so one
// base pair + offsets 0/16/32/48 moves a 64B row without a VGPR round-trip.
// ---------------------------------------------------------------------------
__device__ __forceinline__ void gemm128_core(const unsigned short* __restrict__ A,
                                             const unsigned short* __restrict__ Bt,
                                             int mBase, int nBase, int Kdim,
                                             unsigned short (&As)[128][32],
                                             unsigned short (&Bs)[128][32],
                                             v8f (&acc)[4][4]) {
  const int tid  = threadIdx.x;
  const int wave = tid >> 5;
  const int wm = (wave >> 1) * 64;
  const int wn = (wave & 1) * 64;

  for (int k0 = 0; k0 < Kdim; k0 += 32) {
#if ASYNC_LDS
    {
      gv4i_p ga = (gv4i_p)(A + (size_t)(mBase + tid) * Kdim + k0);
      lv4i_p sa = (lv4i_p)&As[tid][0];
      __builtin_amdgcn_global_load_async_to_lds_b128(ga, sa, 0, 0);
      __builtin_amdgcn_global_load_async_to_lds_b128(ga, sa, 16, 0);
      __builtin_amdgcn_global_load_async_to_lds_b128(ga, sa, 32, 0);
      __builtin_amdgcn_global_load_async_to_lds_b128(ga, sa, 48, 0);
      gv4i_p gb = (gv4i_p)(Bt + (size_t)(nBase + tid) * Kdim + k0);
      lv4i_p sb = (lv4i_p)&Bs[tid][0];
      __builtin_amdgcn_global_load_async_to_lds_b128(gb, sb, 0, 0);
      __builtin_amdgcn_global_load_async_to_lds_b128(gb, sb, 16, 0);
      __builtin_amdgcn_global_load_async_to_lds_b128(gb, sb, 32, 0);
      __builtin_amdgcn_global_load_async_to_lds_b128(gb, sb, 48, 0);
      asm volatile("s_wait_asynccnt 0" ::: "memory");
    }
#else
    // cooperative stage: thread t copies one 32-element row of each tile
    const u32x4* ga = reinterpret_cast<const u32x4*>(A + (size_t)(mBase + tid) * Kdim + k0);
    u32x4* sa = reinterpret_cast<u32x4*>(&As[tid][0]);
    sa[0] = ga[0]; sa[1] = ga[1]; sa[2] = ga[2]; sa[3] = ga[3];
    const u32x4* gb = reinterpret_cast<const u32x4*>(Bt + (size_t)(nBase + tid) * Kdim + k0);
    u32x4* sb = reinterpret_cast<u32x4*>(&Bs[tid][0]);
    sb[0] = gb[0]; sb[1] = gb[1]; sb[2] = gb[2]; sb[3] = gb[3];
#endif
    if (k0 + 32 < Kdim) {   // prefetch next K tile -> global_prefetch_b8
      __builtin_prefetch(A  + (size_t)(mBase + tid) * Kdim + k0 + 32, 0, 1);
      __builtin_prefetch(Bt + (size_t)(nBase + tid) * Kdim + k0 + 32, 0, 1);
    }
    __syncthreads();

    Frag af[4], bfr[4];
#pragma unroll
    for (int i = 0; i < 4; ++i) load_a_frag(af[i],  &As[wm + 16 * i][0], 32);
#pragma unroll
    for (int j = 0; j < 4; ++j) load_b_frag(bfr[j], &Bs[wn + 16 * j][0], 32);
#pragma unroll
    for (int i = 0; i < 4; ++i)
#pragma unroll
      for (int j = 0; j < 4; ++j)
        acc[i][j] = wmma_bf16(af[i], bfr[j], acc[i][j]);
    __syncthreads();
  }
}

// ---------------------------------------------------------------------------
// QKV GEMM: qkv = Xb @ Wqkv ; scatter epilogue into K/Q ([b,h,t,d]) and
// V transposed ([b,h,d,t]).  Q is pre-scaled by C^-0.5 = 2^-5 (exact in bf16).
// split order (as in reference): cols [0,C)=k, [C,2C)=q, [2C,3C)=v
// ---------------------------------------------------------------------------
__global__ __launch_bounds__(128) void k_gemm_qkv(const unsigned short* __restrict__ Xb,
                                                  const unsigned short* __restrict__ Wt,
                                                  unsigned short* __restrict__ Qb,
                                                  unsigned short* __restrict__ Kb,
                                                  unsigned short* __restrict__ Vt) {
  __shared__ __align__(16) unsigned short As[128][32];
  __shared__ __align__(16) unsigned short Bs[128][32];
  v8f acc[4][4] = {};
  const int mBase = blockIdx.y * 128;
  const int nBase = blockIdx.x * 128;
  gemm128_core(Xb, Wt, mBase, nBase, C_, As, Bs, acc);

  const int tid  = threadIdx.x;
  const int wave = tid >> 5;
  const int wm = (wave >> 1) * 64, wn = (wave & 1) * 64;
  const int lane = tid & 31, hi = lane >> 4, n15 = lane & 15;
#pragma unroll
  for (int i = 0; i < 4; ++i)
#pragma unroll
    for (int j = 0; j < 4; ++j)
#pragma unroll
      for (int r = 0; r < 8; ++r) {
        const int row = mBase + wm + 16 * i + r + hi * 8;   // token index
        const int col = nBase + wn + 16 * j + n15;          // qkv column
        const int b = row >> 11, t = row & (T_ - 1);
        const int which = col >> 10, c = col & (C_ - 1);
        const int h = c >> 6, d = c & 63;
        const float f = acc[i][j][r];
        const size_t bh = (size_t)(b * H_ + h);
        if (which == 0)      Kb[(bh * T_ + t) * D_ + d] = f2bf(f);
        else if (which == 1) Qb[(bh * T_ + t) * D_ + d] = f2bf(f * 0.03125f);
        else                 Vt[(bh * D_ + d) * T_ + t] = f2bf(f);
      }
}

// ---------------------------------------------------------------------------
// Flash attention: one workgroup = 4 waves, each wave = 16 query rows.
// Online softmax over key blocks of 32.  S: 4 WMMA, P@V: 4 WMMA per block.
// ---------------------------------------------------------------------------
__global__ __launch_bounds__(128) void k_attn(const unsigned short* __restrict__ Qb,
                                              const unsigned short* __restrict__ Kb,
                                              const unsigned short* __restrict__ Vt,
                                              unsigned short* __restrict__ AO) {
  __shared__ __align__(16) unsigned short Pls[4][16 * 32];  // wave-private P staging
  const int wave = threadIdx.x >> 5;
  const int lane = threadIdx.x & 31;
  const int hi = lane >> 4, n15 = lane & 15;
  const int b = blockIdx.z, h = blockIdx.y;
  const int qBase = blockIdx.x * 64 + wave * 16;
  const size_t bh = (size_t)(b * H_ + h);
  const unsigned short* Qp = Qb + bh * T_ * D_;
  const unsigned short* Kp = Kb + bh * T_ * D_;
  const unsigned short* Vp = Vt + bh * D_ * T_;

  Frag qf0, qf1;                                    // Q tile 16x64, 2 K-chunks
  load_a_frag(qf0, Qp + (size_t)qBase * D_ + 0,  D_);
  load_a_frag(qf1, Qp + (size_t)qBase * D_ + 32, D_);

  float mrow[8], lrow[8];
  v8f o[4] = {};
#pragma unroll
  for (int r = 0; r < 8; ++r) { mrow[r] = -3.0e38f; lrow[r] = 0.f; }

  const int kbMax = (qBase + 15) >> 5;              // causal: keys <= qBase+15
  for (int kb = 0; kb <= kbMax; ++kb) {
    const int j0 = kb * 32;
    v8f s0 = {}, s1 = {};
    {
      Frag kf;
      load_b_frag(kf, Kp + (size_t)(j0 +  0) * D_ +  0, D_); s0 = wmma_bf16(qf0, kf, s0);
      load_b_frag(kf, Kp + (size_t)(j0 +  0) * D_ + 32, D_); s0 = wmma_bf16(qf1, kf, s0);
      load_b_frag(kf, Kp + (size_t)(j0 + 16) * D_ +  0, D_); s1 = wmma_bf16(qf0, kf, s1);
      load_b_frag(kf, Kp + (size_t)(j0 + 16) * D_ + 32, D_); s1 = wmma_bf16(qf1, kf, s1);
    }
#pragma unroll
    for (int r = 0; r < 8; ++r) {
      const int qi = qBase + r + hi * 8;
      float e0 = ((j0 + n15)      <= qi) ? s0[r] : -3.0e38f;   // causal mask
      float e1 = ((j0 + 16 + n15) <= qi) ? s1[r] : -3.0e38f;
      float v = fmaxf(e0, e1);                     // row max across 16 lanes
#pragma unroll
      for (int off = 8; off; off >>= 1) v = fmaxf(v, __shfl_xor(v, off, 32));
      const float mn = fmaxf(mrow[r], v);
      const float al = __expf(mrow[r] - mn);
      e0 = __expf(e0 - mn);
      e1 = __expf(e1 - mn);
      float sum = e0 + e1;
#pragma unroll
      for (int off = 8; off; off >>= 1) sum += __shfl_xor(sum, off, 32);
      lrow[r] = lrow[r] * al + sum;
      mrow[r] = mn;
#pragma unroll
      for (int dn = 0; dn < 4; ++dn) o[dn][r] *= al;
      const int prow = r + hi * 8;                 // C-layout -> LDS row-major P
      Pls[wave][prow * 32 + n15]      = f2bf(e0);
      Pls[wave][prow * 32 + 16 + n15] = f2bf(e1);
    }
    asm volatile("s_wait_dscnt 0" ::: "memory");   // wave-private LDS RAW
    Frag pf;
    load_a_frag(pf, &Pls[wave][0], 32);            // P as A-fragment (16x32)
#pragma unroll
    for (int dn = 0; dn < 4; ++dn) {
      Frag vf;                                     // Vt rows: [d][t], contiguous K
      load_b_frag(vf, Vp + (size_t)(dn * 16) * T_ + j0, T_);
      o[dn] = wmma_bf16(pf, vf, o[dn]);
    }
  }

#pragma unroll
  for (int dn = 0; dn < 4; ++dn)
#pragma unroll
    for (int r = 0; r < 8; ++r) {
      const int row = b * T_ + qBase + r + hi * 8;
      const int col = h * D_ + dn * 16 + n15;
      AO[(size_t)row * C_ + col] = f2bf(o[dn][r] / lrow[r]);
    }
}

// ---------------------------------------------------------------------------
// Output projection: out(fp32) = AO(bf16) @ Wproj
// ---------------------------------------------------------------------------
__global__ __launch_bounds__(128) void k_gemm_proj(const unsigned short* __restrict__ Ab,
                                                   const unsigned short* __restrict__ Wt,
                                                   float* __restrict__ out) {
  __shared__ __align__(16) unsigned short As[128][32];
  __shared__ __align__(16) unsigned short Bs[128][32];
  v8f acc[4][4] = {};
  const int mBase = blockIdx.y * 128;
  const int nBase = blockIdx.x * 128;
  gemm128_core(Ab, Wt, mBase, nBase, C_, As, Bs, acc);

  const int tid  = threadIdx.x;
  const int wave = tid >> 5;
  const int wm = (wave >> 1) * 64, wn = (wave & 1) * 64;
  const int lane = tid & 31, hi = lane >> 4, n15 = lane & 15;
#pragma unroll
  for (int i = 0; i < 4; ++i)
#pragma unroll
    for (int j = 0; j < 4; ++j)
#pragma unroll
      for (int r = 0; r < 8; ++r) {
        const int row = mBase + wm + 16 * i + r + hi * 8;
        const int col = nBase + wn + 16 * j + n15;
        out[(size_t)row * C_ + col] = acc[i][j][r];
      }
}

// ---------------------------------------------------------------------------
// Host launcher
// ---------------------------------------------------------------------------
extern "C" void kernel_launch(void* const* d_in, const int* in_sizes, int n_in,
                              void* d_out, int out_size, void* d_ws, size_t ws_size,
                              hipStream_t stream) {
  (void)in_sizes; (void)n_in; (void)out_size; (void)ws_size;
  const float* x     = (const float*)d_in[0];   // [B,T,C]
  const float* Wqkv  = (const float*)d_in[1];   // [C,3C]
  const float* Wproj = (const float*)d_in[2];   // [C,C]
  float* out = (float*)d_out;                   // [B,T,C] fp32

  char* ws = (char*)d_ws;
  size_t off = 0;
  auto carve = [&](size_t bytes) {
    void* p = ws + off;
    off += (bytes + 255) & ~(size_t)255;
    return p;
  };
  unsigned short* Xb  = (unsigned short*)carve((size_t)NTOK * C_ * 2);  // x bf16
  unsigned short* WqT = (unsigned short*)carve((size_t)N3C * C_ * 2);   // Wqkv^T bf16
  unsigned short* WpT = (unsigned short*)carve((size_t)C_ * C_ * 2);    // Wproj^T bf16
  unsigned short* Qb  = (unsigned short*)carve((size_t)NTOK * C_ * 2);  // [b,h,t,d]
  unsigned short* Kb  = (unsigned short*)carve((size_t)NTOK * C_ * 2);  // [b,h,t,d]
  unsigned short* Vt  = (unsigned short*)carve((size_t)NTOK * C_ * 2);  // [b,h,d,t]
  unsigned short* AO  = (unsigned short*)carve((size_t)NTOK * C_ * 2);  // attn out bf16

  k_cast<<<(NTOK * C_ + 255) / 256, 256, 0, stream>>>(x, Xb, NTOK * C_);
  k_transpose_cast<<<dim3(N3C / 16, C_ / 16), dim3(16, 16), 0, stream>>>(Wqkv, WqT, C_, N3C);
  k_transpose_cast<<<dim3(C_ / 16, C_ / 16), dim3(16, 16), 0, stream>>>(Wproj, WpT, C_, C_);

  k_gemm_qkv<<<dim3(N3C / 128, NTOK / 128), 128, 0, stream>>>(Xb, WqT, Qb, Kb, Vt);
  k_attn<<<dim3(T_ / 64, H_, B_), 128, 0, stream>>>(Qb, Kb, Vt, AO);
  k_gemm_proj<<<dim3(C_ / 128, NTOK / 128), 128, 0, stream>>>(AO, WpT, out);
}
